// TrackBlock_13932873908953
// MI455X (gfx1250) — compile-verified, matrix operand order checked by reading, not run on hardware
//
#include <hip/hip_runtime.h>

// ---------------------------------------------------------------------------
// CDNA5 (gfx1250) wave32 WMMA building blocks
// ---------------------------------------------------------------------------
typedef __attribute__((ext_vector_type(16))) __bf16 bf16x16;
typedef __attribute__((ext_vector_type(8)))  float  f32x8;

#define LSEQ  192
#define DNODE 256
#define DEDGE 128
#define DHEAD 48
#define DPADH 64    // padded per-head stride for Q/K
#define NHEAD 8
#define HDIM  384   // NHEAD*DHEAD
#define HPAD  512   // NHEAD*DPADH

__device__ __forceinline__ __bf16 bf_zero() {
  return __builtin_bit_cast(__bf16, (unsigned short)0);
}
__device__ __forceinline__ __bf16 to_bf(__bf16 v) { return v; }
__device__ __forceinline__ __bf16 to_bf(float f) {
  unsigned u = __builtin_bit_cast(unsigned, f);
  u += 0x7fffu + ((u >> 16) & 1u);               // round-to-nearest-even
  return __builtin_bit_cast(__bf16, (unsigned short)(u >> 16));
}
__device__ __forceinline__ float bf2f(__bf16 h) {
  unsigned u = ((unsigned)__builtin_bit_cast(unsigned short, h)) << 16;
  return __builtin_bit_cast(float, u);
}
// Unguarded fragment load, K-contiguous source. Per-lane layout for 16x16x32:
// elems 0..7 -> K = g*8 + 0..7, elems 8..15 -> K = 16 + g*8 + 0..7.
template <typename T>
__device__ __forceinline__ bf16x16 frag_ld(const T* p, int g) {
  const T* q = p + g * 8;
  bf16x16 f;
#pragma unroll
  for (int e = 0; e < 8; ++e) f[e] = to_bf(q[e]);
#pragma unroll
  for (int e = 0; e < 8; ++e) f[8 + e] = to_bf(q[16 + e]);
  return f;
}
__device__ __forceinline__ f32x8 wmma_bf16(bf16x16 a, bf16x16 b, f32x8 c) {
  return __builtin_amdgcn_wmma_f32_16x16x32_bf16(false, a, false, b, (short)0, c, false, false);
}

// ---------------------------------------------------------------------------
// GEMM with pre-transposed bf16 B:  C[M,N] = epi(alpha * A[M,K] x BT[N,K]^T)
//   ABF: A element type (0=f32, 1=bf16).  K must be a multiple of 32.
//   opc: 0 store, 1 add residual, 2 C *= sigmoid(acc);  act: 1 -> relu
//   c_trans: store C transposed as CT[col*ldc + row]
// One wave per 32x32 C tile (2x2 accumulators: 4 WMMA per 4 fragment loads).
// Branch-free inner loop: OOB rows/cols clamped for loads; stores guarded.
// ---------------------------------------------------------------------------
template <int ABF>
__global__ __launch_bounds__(32)
void gemm_bt_kernel(const void* Av, long lda, long bsA,
                    const __bf16* BT, long ldb, long bsB,
                    void* Cv, int c_bf16, int c_trans, long ldc, long bsC,
                    const float* bias, const float* res, long ldr, long bsR,
                    int M, int N, int K, float alpha, int opc, int act) {
  int lane = threadIdx.x, sub = lane & 15, g = lane >> 4;
  long b = blockIdx.z;
  int rbase = blockIdx.x * 32, cbase = blockIdx.y * 32;
  int row0 = rbase + sub,      row1 = row0 + 16;
  int col0 = cbase + sub,      col1 = col0 + 16;
  int r0c = row0 < M ? row0 : M - 1;   // clamp: loads always valid,
  int r1c = row1 < M ? row1 : M - 1;   // bogus results never stored
  int c0c = col0 < N ? col0 : N - 1;
  int c1c = col1 < N ? col1 : N - 1;
  const float*  Af0 = (const float*)Av  + b * bsA + (long)r0c * lda;
  const float*  Af1 = (const float*)Av  + b * bsA + (long)r1c * lda;
  const __bf16* Ah0 = (const __bf16*)Av + b * bsA + (long)r0c * lda;
  const __bf16* Ah1 = (const __bf16*)Av + b * bsA + (long)r1c * lda;
  const __bf16* Bp0 = BT + b * bsB + (long)c0c * ldb;
  const __bf16* Bp1 = BT + b * bsB + (long)c1c * ldb;
  f32x8 acc[2][2] = {};
  for (int k0 = 0; k0 < K; k0 += 32) {
    bf16x16 a0 = ABF ? frag_ld(Ah0 + k0, g) : frag_ld(Af0 + k0, g);
    bf16x16 a1 = ABF ? frag_ld(Ah1 + k0, g) : frag_ld(Af1 + k0, g);
    bf16x16 b0 = frag_ld(Bp0 + k0, g);
    bf16x16 b1 = frag_ld(Bp1 + k0, g);
    acc[0][0] = wmma_bf16(a0, b0, acc[0][0]);
    acc[0][1] = wmma_bf16(a0, b1, acc[0][1]);
    acc[1][0] = wmma_bf16(a1, b0, acc[1][0]);
    acc[1][1] = wmma_bf16(a1, b1, acc[1][1]);
  }
#pragma unroll
  for (int ti = 0; ti < 2; ++ti) {
#pragma unroll
    for (int tj = 0; tj < 2; ++tj) {
#pragma unroll
      for (int v = 0; v < 8; ++v) {
        int r = rbase + ti * 16 + g * 8 + v;
        int c = cbase + tj * 16 + sub;
        if (r < M && c < N) {
          long ci = c_trans ? (b * bsC + (long)c * ldc + r)
                            : (b * bsC + (long)r * ldc + c);
          float val = alpha * acc[ti][tj][v] + (bias ? bias[c] : 0.0f);
          if (opc == 2) {
            float gate = 1.0f / (1.0f + __expf(-val));
            if (c_bf16) { __bf16* C = (__bf16*)Cv; C[ci] = to_bf(bf2f(C[ci]) * gate); }
            else        { float*  C = (float*)Cv;  C[ci] *= gate; }
          } else {
            if (act == 1) val = val > 0.0f ? val : 0.0f;
            if (opc == 1) val += res[b * bsR + (long)r * ldr + c];
            if (c_bf16) ((__bf16*)Cv)[ci] = to_bf(val);
            else        ((float*)Cv)[ci]  = val;
          }
        }
      }
    }
  }
}

// ---------------------------------------------------------------------------
// Fused attention. Q/K: bf16 row-major, HEAD-PADDED (ldq=HPAD, head stride 64,
// cols 48..63 zero -> unguarded K loop). VT: bf16 transposed [HDIM][LSEQ].
// logits = post*(scale*Q.K^T + bias[j,k,h]); softmax(k) in LDS; O = attn x V.
// Phase 1 uses 32x32 macro-tiles (2x2 WMMA), phase 3 32x16 (2x1, shared V frag).
// grid = (NHEAD, outer_batch); block = 256 (8 waves); LDS = 192*192*4.
// ---------------------------------------------------------------------------
__global__ __launch_bounds__(256)
void attn_fused_kernel(const __bf16* Q, const __bf16* Km, const __bf16* VT,
                       long bsQ, long bsVT,
                       const float* bias, float scale, float post_scale,
                       __bf16* O, long ldo, long bsO,
                       float* attn_out) {
  const int Lk = LSEQ;
  int h = blockIdx.x;
  long ob = blockIdx.y;
  int tid = threadIdx.x, w = tid >> 5, lane = tid & 31;
  int sub = lane & 15, g = lane >> 4;
  extern __shared__ float slog[];      // 192*192 floats
  long qoff = ob * bsQ;
  int hq = h * DPADH;                  // padded Q/K head offset
  int hv = h * DHEAD;                  // V/O head offset

  // Phase 1: 6x6 macro-tiles of 32x32, K-dim = 64 (padded, zeros beyond 48)
  for (int t = w; t < 36; t += 8) {
    int j0 = (t / 6) * 32, k0c = (t % 6) * 32;
    const __bf16* qp0 = Q + qoff + (long)(j0 + sub) * HPAD + hq;
    const __bf16* qp1 = qp0 + 16 * HPAD;
    const __bf16* kp0 = Km + qoff + (long)(k0c + sub) * HPAD + hq;
    const __bf16* kp1 = kp0 + 16 * HPAD;
    f32x8 acc[2][2] = {};
#pragma unroll
    for (int kk0 = 0; kk0 < DPADH; kk0 += 32) {
      bf16x16 a0 = frag_ld(qp0 + kk0, g), a1 = frag_ld(qp1 + kk0, g);
      bf16x16 b0 = frag_ld(kp0 + kk0, g), b1 = frag_ld(kp1 + kk0, g);
      acc[0][0] = wmma_bf16(a0, b0, acc[0][0]);
      acc[0][1] = wmma_bf16(a0, b1, acc[0][1]);
      acc[1][0] = wmma_bf16(a1, b0, acc[1][0]);
      acc[1][1] = wmma_bf16(a1, b1, acc[1][1]);
    }
#pragma unroll
    for (int ti = 0; ti < 2; ++ti)
#pragma unroll
      for (int tj = 0; tj < 2; ++tj)
#pragma unroll
        for (int v = 0; v < 8; ++v) {
          int j = j0 + ti * 16 + g * 8 + v, k = k0c + tj * 16 + sub;
          float bi = bias[((long)j * Lk + k) * NHEAD + h];
          slog[j * Lk + k] = post_scale * (scale * acc[ti][tj][v] + bi);
        }
  }
  __syncthreads();
  // Phase 2: softmax over k, one thread per row
  if (tid < Lk) {
    float* rp = slog + tid * Lk;
    float m = -1e30f;
    for (int k = 0; k < Lk; ++k) m = fmaxf(m, rp[k]);
    float s = 0.0f;
    for (int k = 0; k < Lk; ++k) { float e = __expf(rp[k] - m); rp[k] = e; s += e; }
    float inv = 1.0f / s;
    for (int k = 0; k < Lk; ++k) rp[k] *= inv;
  }
  __syncthreads();
  // Phase 3: O = attn x V  (6x3 macro-tiles of 32x16, K = 192, shared V frag)
  for (int t = w; t < 18; t += 8) {
    int j0 = (t / 3) * 32, d0 = (t % 3) * 16;
    const float* apl0 = slog + (long)(j0 + sub) * Lk;
    const float* apl1 = apl0 + 16 * Lk;
    const __bf16* vtp = VT + ob * bsVT + (long)(hv + d0 + sub) * Lk;
    f32x8 acc0 = {}, acc1 = {};
    for (int kk0 = 0; kk0 < Lk; kk0 += 32) {
      bf16x16 bv = frag_ld(vtp + kk0, g);
      acc0 = wmma_bf16(frag_ld(apl0 + kk0, g), bv, acc0);
      acc1 = wmma_bf16(frag_ld(apl1 + kk0, g), bv, acc1);
    }
#pragma unroll
    for (int v = 0; v < 8; ++v) {
      int d = d0 + sub;
      int ja = j0 + g * 8 + v, jb = ja + 16;
      O[ob * bsO + (long)ja * ldo + hv + d] = to_bf(acc0[v]);
      O[ob * bsO + (long)jb * ldo + hv + d] = to_bf(acc1[v]);
    }
  }
  if (attn_out) {                      // IPA attn dump, layout (h, q, k)
    __syncthreads();
    float* dst = attn_out + (long)h * Lk * Lk;
    for (int idx = tid; idx < Lk * Lk; idx += 256) dst[idx] = slog[idx];
  }
}

// ---------------------------------------------------------------------------
// Fused outer-product GEMM: edge1[nm,c] = edge[nm,c] + bo[c]
//        + sum_{i,j} l[n,i]*r[m,j]*WoT[c, i*24+j]   (A synthesized on the fly)
// ---------------------------------------------------------------------------
__global__ __launch_bounds__(32)
void outprod_kernel(const float* lb, const float* rb, const __bf16* WoT,
                    const float* bo, const float* ein, float* eout) {
  int lane = threadIdx.x, sub = lane & 15, g = lane >> 4;
  long nm = (long)blockIdx.x * 16 + sub;
  int n = (int)(nm / LSEQ), m = (int)(nm % LSEQ);
  const float* ln_ = lb + n * 24;
  const float* rm_ = rb + m * 24;
  int c0 = blockIdx.y * 16;
  const __bf16* wp = WoT + (long)(c0 + sub) * 576;
  f32x8 acc = {};
  for (int k0 = 0; k0 < 576; k0 += 32) {
    int kb = k0 + g * 8;
    bf16x16 af;
#pragma unroll
    for (int e = 0; e < 8; ++e) {
      int kk = kb + e;
      af[e] = to_bf(ln_[kk / 24] * rm_[kk % 24]);
    }
#pragma unroll
    for (int e = 0; e < 8; ++e) {
      int kk = kb + 16 + e;
      af[8 + e] = to_bf(ln_[kk / 24] * rm_[kk % 24]);
    }
    acc = wmma_bf16(af, frag_ld(wp + k0, g), acc);
  }
  long base = (long)blockIdx.x * 16;
#pragma unroll
  for (int v = 0; v < 8; ++v) {
    long r = base + g * 8 + v;
    int c = c0 + sub;
    eout[r * DEDGE + c] = ein[r * DEDGE + c] + acc[v] + bo[c];
  }
}

// ------------------------- transposes (bf16 output) ------------------------
__global__ void transpose_w_kernel(const float* in, __bf16* out, int K, int N) {
  long idx = (long)blockIdx.x * blockDim.x + threadIdx.x;
  if (idx >= (long)K * N) return;
  int k = (int)(idx / N), n = (int)(idx % N);
  out[(long)n * K + k] = to_bf(in[idx]);
}
__global__ void transpose_wqk_pad_kernel(const float* in, __bf16* out, int K) {
  long idx = (long)blockIdx.x * blockDim.x + threadIdx.x;
  if (idx >= (long)HPAD * K) return;
  int n = (int)(idx / K), k = (int)(idx % K);
  int h = n >> 6, c = n & 63;
  float v = (c < DHEAD) ? in[(long)k * HDIM + h * DHEAD + c] : 0.0f;
  out[idx] = to_bf(v);
}
__global__ void transpose_et_kernel(const float* in, __bf16* out) {
  long idx = (long)blockIdx.x * blockDim.x + threadIdx.x;
  if (idx >= (long)LSEQ * LSEQ * DEDGE) return;
  long q = idx / ((long)LSEQ * DEDGE);
  long rem = idx % ((long)LSEQ * DEDGE);
  int k = (int)(rem / DEDGE), c = (int)(rem % DEDGE);
  out[q * (long)LSEQ * DEDGE + (long)c * LSEQ + k] = to_bf(in[idx]);
}

// --------------------------- small VALU kernels ----------------------------
__global__ void layernorm_kernel(const float* x, const float* g, const float* b,
                                 __bf16* y, long rows, int C) {
  long r = (long)blockIdx.x * blockDim.x + threadIdx.x;
  if (r >= rows) return;
  const float* xp = x + r * C;
  float mu = 0.0f;
  for (int i = 0; i < C; ++i) mu += xp[i];
  mu /= (float)C;
  float var = 0.0f;
  for (int i = 0; i < C; ++i) { float d = xp[i] - mu; var += d * d; }
  var /= (float)C;
  float inv = rsqrtf(var + 1e-5f);
  __bf16* yp = y + r * C;
  for (int i = 0; i < C; ++i) yp[i] = to_bf((xp[i] - mu) * inv * g[i] + b[i]);
}

__global__ void rbf_kernel(const float* aff, float* out) {
  long jk = (long)blockIdx.x * blockDim.x + threadIdx.x;
  if (jk >= (long)LSEQ * LSEQ) return;
  int j = (int)(jk / LSEQ), k = (int)(jk % LSEQ);
  float s = 1e-12f;
  for (int i = 0; i < 3; ++i) {
    float d = aff[j * 12 + i * 4 + 3] - aff[k * 12 + i * 4 + 3];
    s += d * d;
  }
  float dist = sqrtf(s);
  float* o = out + jk * 64;
  for (int i = 0; i < 64; ++i) {
    float e = (dist - 0.5f * (float)i) * 2.0f;
    o[i] = __expf(-e * e);
  }
}

__global__ void bias_na_kernel(const float* edge, const float* rbf,
                               const float* WbE, const float* WbR, float* out) {
  long jk = (long)blockIdx.x * blockDim.x + threadIdx.x;
  if (jk >= (long)LSEQ * LSEQ) return;
  float a[NHEAD];
#pragma unroll
  for (int h = 0; h < NHEAD; ++h) a[h] = 0.0f;
  const float* ep = edge + jk * DEDGE;
  for (int d = 0; d < DEDGE; ++d) {
    float e = ep[d];
#pragma unroll
    for (int h = 0; h < NHEAD; ++h) a[h] += e * WbE[d * NHEAD + h];
  }
  const float* rp = rbf + jk * 64;
  for (int d = 0; d < 64; ++d) {
    float e = rp[d];
#pragma unroll
    for (int h = 0; h < NHEAD; ++h) a[h] += e * WbR[d * NHEAD + h];
  }
#pragma unroll
  for (int h = 0; h < NHEAD; ++h) out[jk * NHEAD + h] = a[h];
}

__global__ void bias_ea_kernel(const float* rbf, const float* aff,
                               const float* WbR, const float* WbP, float* out) {
  long jk = (long)blockIdx.x * blockDim.x + threadIdx.x;
  if (jk >= (long)LSEQ * LSEQ) return;
  int j = (int)(jk / LSEQ), k = (int)(jk % LSEQ);
  float a[NHEAD];
#pragma unroll
  for (int h = 0; h < NHEAD; ++h) a[h] = 0.0f;
  const float* rp = rbf + jk * 64;
  for (int d = 0; d < 64; ++d) {
    float e = rp[d];
#pragma unroll
    for (int h = 0; h < NHEAD; ++h) a[h] += e * WbR[d * NHEAD + h];
  }
  for (int c = 0; c < 3; ++c) {
    float dif = aff[j * 12 + c * 4 + 3] - aff[k * 12 + c * 4 + 3];
#pragma unroll
    for (int h = 0; h < NHEAD; ++h) a[h] += dif * WbP[c * NHEAD + h];
  }
#pragma unroll
  for (int h = 0; h < NHEAD; ++h) out[jk * NHEAD + h] = a[h];
}

__global__ void to_global_kernel(const float* aff, const float* qp, const float* kp,
                                 const float* vp, float* qpg, float* kpg, float* vpg) {
  int tid = blockIdx.x * blockDim.x + threadIdx.x;
  if (tid >= LSEQ * 128) return;
  int l = tid >> 7, s = tid & 127;
  const float* in; float* out; int off, ncols;
  if (s < 32)      { in = qp; out = qpg; off = s * 3;        ncols = 96;  }
  else if (s < 64) { in = kp; out = kpg; off = (s - 32) * 3; ncols = 96;  }
  else             { in = vp; out = vpg; off = (s - 64) * 3; ncols = 288; }
  const float* R = aff + l * 12;
  const float* v = in + (long)l * ncols + off;
  float x = v[0], y = v[1], z = v[2];
  float* o = out + (long)l * ncols + off;
  o[0] = R[0] * x + R[1] * y + R[2] * z + R[3];
  o[1] = R[4] * x + R[5] * y + R[6] * z + R[7];
  o[2] = R[8] * x + R[9] * y + R[10] * z + R[11];
}

__global__ void ptatt_kernel(const float* qpg, const float* kpg,
                             const float* gamma, float* bias3) {
  long idx = (long)blockIdx.x * blockDim.x + threadIdx.x;
  if (idx >= (long)LSEQ * LSEQ * NHEAD) return;
  int h = (int)(idx & (NHEAD - 1));
  long qk = idx >> 3;
  int q = (int)(qk / LSEQ), k = (int)(qk % LSEQ);
  const float* qv = qpg + (long)q * 96 + h * 12;
  const float* kv = kpg + (long)k * 96 + h * 12;
  float s = 0.0f;
#pragma unroll
  for (int e = 0; e < 12; ++e) { float d = qv[e] - kv[e]; s += d * d; }
  float hw = logf(1.0f + __expf(gamma[h]));
  const float wc = 0.23570226039551584f;  // sqrt(2/(9*4))
  bias3[idx] -= 0.5f * wc * hw * s;
}

__global__ void opt_kernel(const float* attn, const float* vpg, const float* aff,
                           __bf16* cat) {
  int tid = blockIdx.x * blockDim.x + threadIdx.x;
  if (tid >= LSEQ * 64) return;
  int q = tid >> 6, hp = tid & 63, h = hp >> 3, p = hp & 7;
  const float* arow = attn + ((long)h * LSEQ + q) * LSEQ;
  const float* vb = vpg + h * 24 + p * 3;
  float a0 = 0.0f, a1 = 0.0f, a2 = 0.0f;
  for (int k = 0; k < LSEQ; ++k) {
    float a = arow[k];
    const float* vv = vb + (long)k * 288;
    a0 += a * vv[0]; a1 += a * vv[1]; a2 += a * vv[2];
  }
  const float* R = aff + q * 12;
  float x = a0 - R[3], y = a1 - R[7], z = a2 - R[11];
  float o0 = R[0] * x + R[4] * y + R[8] * z;
  float o1 = R[1] * x + R[5] * y + R[9] * z;
  float o2 = R[2] * x + R[6] * y + R[10] * z;
  __bf16* c = cat + (long)q * 1664;
  c[384 + h * 24 + p * 3 + 0] = to_bf(o0);
  c[384 + h * 24 + p * 3 + 1] = to_bf(o1);
  c[384 + h * 24 + p * 3 + 2] = to_bf(o2);
  c[576 + h * 8 + p] = to_bf(sqrtf(o0 * o0 + o1 * o1 + o2 * o2 + 1e-8f));
}

__global__ void bb_kernel(const float* aff, const float* upd, float* aff_out) {
  int l = blockIdx.x * blockDim.x + threadIdx.x;
  if (l >= LSEQ) return;
  const float* u = upd + l * 6;
  float b = u[0], c = u[1], d = u[2];
  float tx = u[3], ty = u[4], tz = u[5];
  float inv = rsqrtf(1.0f + b * b + c * c + d * d);
  float qw = inv, qx = b * inv, qy = c * inv, qz = d * inv;
  float Ru[9] = {1 - 2 * (qy * qy + qz * qz), 2 * (qx * qy - qw * qz), 2 * (qx * qz + qw * qy),
                 2 * (qx * qy + qw * qz), 1 - 2 * (qx * qx + qz * qz), 2 * (qy * qz - qw * qx),
                 2 * (qx * qz - qw * qy), 2 * (qy * qz + qw * qx), 1 - 2 * (qx * qx + qy * qy)};
  const float* A = aff + l * 12;
  float R[9] = {A[0], A[1], A[2], A[4], A[5], A[6], A[8], A[9], A[10]};
  float t[3] = {A[3], A[7], A[11]};
  float* O = aff_out + l * 12;
  for (int i = 0; i < 3; ++i) {
    for (int j = 0; j < 3; ++j)
      O[i * 4 + j] = R[i * 3] * Ru[j] + R[i * 3 + 1] * Ru[3 + j] + R[i * 3 + 2] * Ru[6 + j];
    O[i * 4 + 3] = R[i * 3] * tx + R[i * 3 + 1] * ty + R[i * 3 + 2] * tz + t[i];
  }
}

// ---------------------------------------------------------------------------
// Host orchestration
// ---------------------------------------------------------------------------
extern "C" void kernel_launch(void* const* d_in, const int* in_sizes, int n_in,
                              void* d_out, int out_size, void* d_ws, size_t ws_size,
                              hipStream_t stream) {
  (void)in_sizes; (void)n_in; (void)out_size; (void)ws_size;
#define INF(i) ((const float*)d_in[(i)])
  const float* node = INF(0);
  const float* edge = INF(1);
  const float* aff  = INF(2);
  const float *na_lng = INF(3),  *na_lnb = INF(4),  *na_Wq = INF(5),  *na_Wk = INF(6);
  const float *na_Wv = INF(7),   *na_Wg = INF(8),   *na_WbE = INF(9), *na_WbR = INF(10);
  const float *na_Wo = INF(11),  *na_bo = INF(12);
  const float *op_lng = INF(13), *op_lnb = INF(14), *op_Wl = INF(15), *op_bl = INF(16);
  const float *op_Wr = INF(17),  *op_br = INF(18),  *op_Wo = INF(19), *op_bo = INF(20);
  const float *ea_lng = INF(21), *ea_lnb = INF(22), *ea_Wq = INF(23), *ea_Wk = INF(24);
  const float *ea_Wv = INF(25),  *ea_Wg = INF(26),  *ea_WbR = INF(27),*ea_WbP = INF(28);
  const float *ea_Wo = INF(29),  *ea_bo = INF(30);
  const float *ip_lng = INF(31), *ip_lnb = INF(32), *ip_Wq = INF(33), *ip_Wk = INF(34);
  const float *ip_Wv = INF(35),  *ip_Wqp = INF(36), *ip_Wkp = INF(37),*ip_Wvp = INF(38);
  const float *ip_Wb = INF(39),  *ip_gamma = INF(40),*ip_Wo = INF(41),*ip_bo = INF(42);
  const float *tr_lng = INF(43), *tr_lnb = INF(44), *tr_W1 = INF(45), *tr_b1 = INF(46);
  const float *tr_W2 = INF(47),  *tr_b2 = INF(48),  *tr_W3 = INF(49), *tr_b3 = INF(50);
  const float *bb_lng = INF(51), *bb_lnb = INF(52), *bb_W = INF(53),  *bb_b = INF(54);

  float* out_node = (float*)d_out;
  float* out_edge = out_node + (long)LSEQ * DNODE;
  float* out_aff  = out_edge + (long)LSEQ * LSEQ * DEDGE;

  char* wsp = (char*)d_ws;
  auto allocB = [&](size_t bytes) -> void* {
    void* p = wsp; wsp += (bytes + 255) & ~(size_t)255; return p;
  };
  auto allocF = [&](size_t n) -> float* { return (float*)allocB(n * 4); };
  auto allocH = [&](size_t n) -> __bf16* { return (__bf16*)allocB(n * 2); };

  const long NN_ = (long)LSEQ * LSEQ;           // 36864
  // fp32 buffers
  float* rbfb   = allocF(NN_ * 64);
  float* nabias = allocF(NN_ * NHEAD);
  float* node1  = allocF((long)LSEQ * DNODE);
  float* lb     = allocF((long)LSEQ * 24);
  float* rb     = allocF((long)LSEQ * 24);
  float* edge1  = allocF(NN_ * DEDGE);
  float* eabias = allocF(NN_ * NHEAD);
  float* qp     = allocF((long)LSEQ * 96);
  float* kp     = allocF((long)LSEQ * 96);
  float* vp     = allocF((long)LSEQ * 288);
  float* qpg    = allocF((long)LSEQ * 96);
  float* kpg    = allocF((long)LSEQ * 96);
  float* vpg    = allocF((long)LSEQ * 288);
  float* bias3  = allocF(NN_ * NHEAD);
  float* attn3  = allocF(NN_ * NHEAD);
  float* node2  = allocF((long)LSEQ * DNODE);
  float* updb   = allocF((long)LSEQ * 6);
  // bf16 activations
  __bf16* xn   = allocH((long)LSEQ * DNODE);
  __bf16* xo   = allocH((long)LSEQ * DNODE);
  __bf16* xe   = allocH(NN_ * DEDGE);
  __bf16* xi   = allocH((long)LSEQ * DNODE);
  __bf16* xt   = allocH((long)LSEQ * DNODE);
  __bf16* xb   = allocH((long)LSEQ * DNODE);
  __bf16* h1   = allocH((long)LSEQ * DNODE);
  __bf16* h2   = allocH((long)LSEQ * DNODE);
  __bf16* nao  = allocH((long)LSEQ * HDIM);
  __bf16* eao  = allocH(NN_ * HDIM);
  __bf16* catb = allocH((long)LSEQ * 1664);
  __bf16* naq  = allocH((long)LSEQ * HPAD);     // head-padded Q/K
  __bf16* nak  = allocH((long)LSEQ * HPAD);
  __bf16* navT = allocH((long)HDIM * LSEQ);
  __bf16* eaq  = allocH(NN_ * HPAD);
  __bf16* eak  = allocH(NN_ * HPAD);
  __bf16* eavT = allocH(NN_ * HDIM);
  __bf16* iq   = allocH((long)LSEQ * HPAD);
  __bf16* ik   = allocH((long)LSEQ * HPAD);
  __bf16* ivT  = allocH((long)HDIM * LSEQ);
  __bf16* ET   = allocH(NN_ * DEDGE);
  // bf16 pre-transposed weights (N x K); Q/K weights head-padded (HPAD x K)
  __bf16 *na_WqT = allocH((long)HPAD * DNODE), *na_WkT = allocH((long)HPAD * DNODE);
  __bf16 *na_WvT = allocH(HDIM * DNODE), *na_WgT = allocH(HDIM * DNODE);
  __bf16 *na_WoT = allocH(DNODE * HDIM);
  __bf16 *op_WlT = allocH(24 * DNODE), *op_WrT = allocH(24 * DNODE);
  __bf16 *op_WoT = allocH(DEDGE * 576);
  __bf16 *ea_WqT = allocH((long)HPAD * DEDGE), *ea_WkT = allocH((long)HPAD * DEDGE);
  __bf16 *ea_WvT = allocH(HDIM * DEDGE), *ea_WgT = allocH(HDIM * DEDGE);
  __bf16 *ea_WoT = allocH(DEDGE * HDIM);
  __bf16 *ip_WqT = allocH((long)HPAD * DNODE), *ip_WkT = allocH((long)HPAD * DNODE);
  __bf16 *ip_WvT = allocH(HDIM * DNODE);
  __bf16 *ip_WqpT = allocH(96 * DNODE), *ip_WkpT = allocH(96 * DNODE);
  __bf16 *ip_WvpT = allocH(288 * DNODE);
  __bf16 *ip_WbT = allocH(NHEAD * DEDGE);
  __bf16 *ip_WoT = allocH(DNODE * 1664);
  __bf16 *tr_W1T = allocH(DNODE * DNODE), *tr_W2T = allocH(DNODE * DNODE);
  __bf16 *tr_W3T = allocH(DNODE * DNODE);
  __bf16 *bb_WT  = allocH(6 * DNODE);

  auto trw = [&](const float* in, __bf16* out, int K, int N) {
    long n = (long)K * N;
    transpose_w_kernel<<<(unsigned)((n + 255) / 256), 256, 0, stream>>>(in, out, K, N);
  };
  auto trwqk = [&](const float* in, __bf16* out, int K) {
    long n = (long)HPAD * K;
    transpose_wqk_pad_kernel<<<(unsigned)((n + 255) / 256), 256, 0, stream>>>(in, out, K);
  };
  auto gemm = [&](const void* A, int abf, long lda, long bsA,
                  const __bf16* BT, long ldb, long bsB,
                  void* C, int cbf, int ctr, long ldc, long bsC,
                  const float* bias, const float* res, long ldr, long bsR,
                  int M, int N, int K, float alpha, int opc, int act, int batches) {
    dim3 gr((M + 31) / 32, (N + 31) / 32, batches);
    if (abf)
      gemm_bt_kernel<1><<<gr, dim3(32), 0, stream>>>(A, lda, bsA, BT, ldb, bsB, C, cbf, ctr,
                                                     ldc, bsC, bias, res, ldr, bsR,
                                                     M, N, K, alpha, opc, act);
    else
      gemm_bt_kernel<0><<<gr, dim3(32), 0, stream>>>(A, lda, bsA, BT, ldb, bsB, C, cbf, ctr,
                                                     ldc, bsC, bias, res, ldr, bsR,
                                                     M, N, K, alpha, opc, act);
  };
  auto ln = [&](const float* x, const float* g, const float* b, __bf16* y, long rows, int C) {
    layernorm_kernel<<<(unsigned)((rows + 255) / 256), 256, 0, stream>>>(x, g, b, y, rows, C);
  };
  const float scale = 0.14433756729740643f;   // 1/sqrt(48)
  const float wl    = 0.5773502691896258f;    // sqrt(1/3)
  const size_t attn_lds = (size_t)LSEQ * LSEQ * 4;

  // ---- weight transposes ----
  trwqk(na_Wq, na_WqT, DNODE);  trwqk(na_Wk, na_WkT, DNODE);
  trw(na_Wv, na_WvT, DNODE, HDIM);  trw(na_Wg, na_WgT, DNODE, HDIM);
  trw(na_Wo, na_WoT, HDIM, DNODE);
  trw(op_Wl, op_WlT, DNODE, 24);    trw(op_Wr, op_WrT, DNODE, 24);
  trw(op_Wo, op_WoT, 576, DEDGE);
  trwqk(ea_Wq, ea_WqT, DEDGE);  trwqk(ea_Wk, ea_WkT, DEDGE);
  trw(ea_Wv, ea_WvT, DEDGE, HDIM);  trw(ea_Wg, ea_WgT, DEDGE, HDIM);
  trw(ea_Wo, ea_WoT, HDIM, DEDGE);
  trwqk(ip_Wq, ip_WqT, DNODE);  trwqk(ip_Wk, ip_WkT, DNODE);
  trw(ip_Wv, ip_WvT, DNODE, HDIM);
  trw(ip_Wqp, ip_WqpT, DNODE, 96);  trw(ip_Wkp, ip_WkpT, DNODE, 96);
  trw(ip_Wvp, ip_WvpT, DNODE, 288);
  trw(ip_Wb, ip_WbT, DEDGE, NHEAD); trw(ip_Wo, ip_WoT, 1664, DNODE);
  trw(tr_W1, tr_W1T, DNODE, DNODE); trw(tr_W2, tr_W2T, DNODE, DNODE);
  trw(tr_W3, tr_W3T, DNODE, DNODE); trw(bb_W, bb_WT, DNODE, 6);

  // ---- geometry features ----
  rbf_kernel<<<144, 256, 0, stream>>>(aff, rbfb);

  // ---- node attention ----
  ln(node, na_lng, na_lnb, xn, LSEQ, DNODE);
  gemm(xn,1,DNODE,0, na_WqT,DNODE,0, naq,1,0,HPAD,0, nullptr,nullptr,0,0, LSEQ,HPAD,DNODE,1.f,0,0,1);
  gemm(xn,1,DNODE,0, na_WkT,DNODE,0, nak,1,0,HPAD,0, nullptr,nullptr,0,0, LSEQ,HPAD,DNODE,1.f,0,0,1);
  gemm(xn,1,DNODE,0, na_WvT,DNODE,0, navT,1,1,LSEQ,0, nullptr,nullptr,0,0, LSEQ,HDIM,DNODE,1.f,0,0,1);
  bias_na_kernel<<<144, 256, 0, stream>>>(edge, rbfb, na_WbE, na_WbR, nabias);
  attn_fused_kernel<<<dim3(NHEAD, 1), 256, attn_lds, stream>>>(
      naq, nak, navT, 0, 0, nabias, scale, 1.0f, nao, HDIM, 0, nullptr);
  gemm(xn,1,DNODE,0, na_WgT,DNODE,0, nao,1,0,HDIM,0, nullptr,nullptr,0,0, LSEQ,HDIM,DNODE,1.f,2,0,1);
  gemm(nao,1,HDIM,0, na_WoT,HDIM,0, node1,0,0,DNODE,0, na_bo, node,DNODE,0, LSEQ,DNODE,HDIM,1.f,1,0,1);

  // ---- outer product edge update ----
  ln(node1, op_lng, op_lnb, xo, LSEQ, DNODE);
  gemm(xo,1,DNODE,0, op_WlT,DNODE,0, lb,0,0,24,0, op_bl,nullptr,0,0, LSEQ,24,DNODE,1.f,0,0,1);
  gemm(xo,1,DNODE,0, op_WrT,DNODE,0, rb,0,0,24,0, op_br,nullptr,0,0, LSEQ,24,DNODE,1.f,0,0,1);
  outprod_kernel<<<dim3(2304, 8), 32, 0, stream>>>(lb, rb, op_WoT, op_bo, edge, edge1);

  // ---- edge attention ----
  ln(edge1, ea_lng, ea_lnb, xe, NN_, DEDGE);
  gemm(xe,1,DEDGE,0, ea_WqT,DEDGE,0, eaq,1,0,HPAD,0, nullptr,nullptr,0,0, (int)NN_,HPAD,DEDGE,1.f,0,0,1);
  gemm(xe,1,DEDGE,0, ea_WkT,DEDGE,0, eak,1,0,HPAD,0, nullptr,nullptr,0,0, (int)NN_,HPAD,DEDGE,1.f,0,0,1);
  gemm(xe,1,DEDGE,(long)LSEQ * DEDGE, ea_WvT,DEDGE,0, eavT,1,1,LSEQ,(long)LSEQ * HDIM,
       nullptr,nullptr,0,0, LSEQ,HDIM,DEDGE,1.f,0,0,LSEQ);
  bias_ea_kernel<<<144, 256, 0, stream>>>(rbfb, aff, ea_WbR, ea_WbP, eabias);
  attn_fused_kernel<<<dim3(NHEAD, LSEQ), 256, attn_lds, stream>>>(
      eaq, eak, eavT, (long)LSEQ * HPAD, (long)LSEQ * HDIM,
      eabias, scale, 1.0f, eao, HDIM, (long)LSEQ * HDIM, nullptr);
  gemm(xe,1,DEDGE,0, ea_WgT,DEDGE,0, eao,1,0,HDIM,0, nullptr,nullptr,0,0, (int)NN_,HDIM,DEDGE,1.f,2,0,1);
  gemm(eao,1,HDIM,0, ea_WoT,HDIM,0, out_edge,0,0,DEDGE,0, ea_bo, edge1,DEDGE,0,
       (int)NN_,DEDGE,HDIM,1.f,1,0,1);

  // ---- IPA ----
  ln(node1, ip_lng, ip_lnb, xi, LSEQ, DNODE);
  gemm(xi,1,DNODE,0, ip_WqT,DNODE,0, iq,1,0,HPAD,0, nullptr,nullptr,0,0, LSEQ,HPAD,DNODE,1.f,0,0,1);
  gemm(xi,1,DNODE,0, ip_WkT,DNODE,0, ik,1,0,HPAD,0, nullptr,nullptr,0,0, LSEQ,HPAD,DNODE,1.f,0,0,1);
  gemm(xi,1,DNODE,0, ip_WvT,DNODE,0, ivT,1,1,LSEQ,0, nullptr,nullptr,0,0, LSEQ,HDIM,DNODE,1.f,0,0,1);
  gemm(xi,1,DNODE,0, ip_WqpT,DNODE,0, qp,0,0,96,0,  nullptr,nullptr,0,0, LSEQ,96,DNODE,1.f,0,0,1);
  gemm(xi,1,DNODE,0, ip_WkpT,DNODE,0, kp,0,0,96,0,  nullptr,nullptr,0,0, LSEQ,96,DNODE,1.f,0,0,1);
  gemm(xi,1,DNODE,0, ip_WvpT,DNODE,0, vp,0,0,288,0, nullptr,nullptr,0,0, LSEQ,288,DNODE,1.f,0,0,1);
  gemm(out_edge,0,DEDGE,0, ip_WbT,DEDGE,0, bias3,0,0,NHEAD,0, nullptr,nullptr,0,0,
       (int)NN_,NHEAD,DEDGE,1.f,0,0,1);
  to_global_kernel<<<96, 256, 0, stream>>>(aff, qp, kp, vp, qpg, kpg, vpg);
  ptatt_kernel<<<1152, 256, 0, stream>>>(qpg, kpg, ip_gamma, bias3);
  attn_fused_kernel<<<dim3(NHEAD, 1), 256, attn_lds, stream>>>(
      iq, ik, ivT, 0, 0, bias3, scale, wl, catb, 1664, 0, attn3);
  opt_kernel<<<48, 256, 0, stream>>>(attn3, vpg, aff, catb);
  transpose_et_kernel<<<(unsigned)((NN_ * DEDGE + 255) / 256), 256, 0, stream>>>(out_edge, ET);
  gemm(attn3,0,NN_,LSEQ, ET,LSEQ,(long)LSEQ * DEDGE, catb + 640,1,0,DEDGE,1664,
       nullptr,nullptr,0,0, NHEAD,DEDGE,LSEQ,1.f,0,0,LSEQ);
  gemm(catb,1,1664,0, ip_WoT,1664,0, node2,0,0,DNODE,0, ip_bo, node1,DNODE,0,
       LSEQ,DNODE,1664,1.f,1,0,1);

  // ---- transition ----
  ln(node2, tr_lng, tr_lnb, xt, LSEQ, DNODE);
  gemm(xt,1,DNODE,0, tr_W1T,DNODE,0, h1,1,0,DNODE,0, tr_b1,nullptr,0,0, LSEQ,DNODE,DNODE,1.f,0,1,1);
  gemm(h1,1,DNODE,0, tr_W2T,DNODE,0, h2,1,0,DNODE,0, tr_b2,nullptr,0,0, LSEQ,DNODE,DNODE,1.f,0,1,1);
  gemm(h2,1,DNODE,0, tr_W3T,DNODE,0, out_node,0,0,DNODE,0, tr_b3, node2,DNODE,0,
       LSEQ,DNODE,DNODE,1.f,1,0,1);

  // ---- backbone update ----
  ln(out_node, bb_lng, bb_lnb, xb, LSEQ, DNODE);
  gemm(xb,1,DNODE,0, bb_WT,DNODE,0, updb,0,0,6,0, bb_b,nullptr,0,0, LSEQ,6,DNODE,1.f,0,0,1);
  bb_kernel<<<1, 256, 0, stream>>>(aff, updb, out_aff);
#undef INF
}
